// AttnEncoder_55241869361880
// MI455X (gfx1250) — matrix operand order repaired and checked by reading, not compiled, not run on hardware
//
#include <hip/hip_runtime.h>
#include <hip/hip_bf16.h>

// CDNA5 (gfx1250) wave32 WMMA kernel for the attention-LSTM encoder.
// D = A(16x32 bf16) x B(32x16 bf16) + C(16x16 f32) via v_wmma_f32_16x16x32_bf16.

typedef __attribute__((ext_vector_type(16))) __bf16 v16bf;
typedef __attribute__((ext_vector_type(8)))  __bf16 v8bf;
typedef __attribute__((ext_vector_type(8)))  float  v8f;

// Explicit global-address-space pointers: keeps laundered weight streams on the
// GLOBAL encoding (LOADcnt only) instead of FLAT (LOADcnt+DScnt).
typedef __bf16 __attribute__((address_space(1))) const* gbp;
typedef v8bf   __attribute__((address_space(1))) const* gv8p;

#define B_ 512
#define T_ 128
#define F_ 128
#define H_ 256
#define G_ 1024   // 4*H
#define BM 16     // batch rows per workgroup
#define GP 20     // gates row padding (floats) -> bank-conflict-free C store/readback

// ---------------- prep kernel 1: x_score[b,f] = sum_t in[b,t,f] * w_x[t] ----------------
__global__ void k_xscore(const float* __restrict__ in, const float* __restrict__ wx,
                         float* __restrict__ xs) {
  int idx = blockIdx.x * blockDim.x + threadIdx.x;   // over B*F, coalesced in f
  int b = idx / F_, f = idx % F_;
  float s = 0.f;
#pragma unroll 4
  for (int t = 0; t < T_; ++t) s = fmaf(in[(size_t)(b * T_ + t) * F_ + f], wx[t], s);
  xs[idx] = s;
}

// ---------------- prep kernel 2: pack W (G_ x K) into WMMA B-fragment layout ------------
// B matrix is W^T (K x N). Packed in this code's contiguous k order (not AMD's table order): lane L holds column n = L%16,
// K = (L/16)*16 + slot, slot = 0..15 ascending (2 bf16 per VGPR dword).
// Block (ntile,ktile) = 512 bf16 = 1 KB; lane's 16 bf16 are 32 contiguous bytes.
__global__ void k_pack(const float* __restrict__ w, __bf16* __restrict__ dst, int K) {
  int idx = blockIdx.x * blockDim.x + threadIdx.x;   // over G_*K
  int n = idx / K, k = idx % K;
  int nt = n >> 4, kt = k >> 5, kl = k & 31;
  int lane = (n & 15) + ((kl >> 4) << 4);
  int slot = kl & 15;
  int nkt = K >> 5;
  dst[(size_t)(nt * nkt + kt) * 512 + lane * 16 + slot] = (__bf16)w[(size_t)n * K + k];
}

__device__ __forceinline__ float sigf(float x) { return 1.f / (1.f + __expf(-x)); }

// ---------------- main persistent recurrence kernel -------------------------------------
__global__ void __launch_bounds__(256)
k_lstm(const float* __restrict__ in, const float* __restrict__ xs,
       const __bf16* __restrict__ wih, const __bf16* __restrict__ whh,
       const float* __restrict__ bih, const float* __restrict__ bhh,
       const float* __restrict__ h0, const float* __restrict__ c0,
       float* __restrict__ attns, float* __restrict__ enc) {
  // Row strides padded: hb/wib +8 bf16 (16B) rotates 4 banks/row; gates +20 floats
  // puts the two C-fragment M-halves on disjoint bank groups.
  __shared__ __bf16 hb[BM][H_ + 8];        // h state, bf16, A-operand source
  __shared__ __bf16 wib[BM][F_ + 8];       // wi = a*x_t, bf16, A-operand source
  __shared__ float  gates[BM][G_ + GP];    // staged GEMM output

  const int tid = threadIdx.x;
  const int b   = tid >> 4;                // 0..15 : batch row within tile
  const int l   = tid & 15;                // 0..15 : column-group lane
  const int bg  = blockIdx.x * BM + b;     // global batch row
  const int wv  = tid >> 5;                // wave 0..7 : owns gate cols [wv*128, wv*128+128)
  const int ln  = tid & 31;                // lane within wave

  // ---- softmax(x_score) once: shift-invariance makes it time-invariant ----
  float areg[8];
  {
    const float* xsp = xs + (size_t)bg * F_ + l * 8;
    float x[8], m = -INFINITY;
#pragma unroll
    for (int i = 0; i < 8; ++i) { x[i] = xsp[i]; m = fmaxf(m, x[i]); }
    for (int off = 8; off > 0; off >>= 1) m = fmaxf(m, __shfl_xor(m, off, 16));
    float s = 0.f;
#pragma unroll
    for (int i = 0; i < 8; ++i) { x[i] = __expf(x[i] - m); s += x[i]; }
    for (int off = 8; off > 0; off >>= 1) s += __shfl_xor(s, off, 16);
    float r = 1.f / s;
#pragma unroll
    for (int i = 0; i < 8; ++i) areg[i] = x[i] * r;
  }

  // ---- persistent state: c in registers, h (bf16) in LDS ----
  float creg[16];
#pragma unroll
  for (int i = 0; i < 16; ++i) {
    int j = l + 16 * i;
    creg[i] = c0[(size_t)bg * H_ + j];
    hb[b][j] = (__bf16)h0[(size_t)bg * H_ + j];
  }

  // hoist per-wave bias (gate columns fixed across t)
  float biasr[8];
#pragma unroll
  for (int nt = 0; nt < 8; ++nt) {
    int n = wv * 128 + nt * 16 + (ln & 15);
    biasr[nt] = bih[n] + bhh[n];
  }
  __syncthreads();

  const int row = ln & 15, sel = ln >> 4;

  for (int t = 0; t < T_; ++t) {
    // ---- phase 1: wi = a * x_t (bf16 into LDS); replay attns ----
    {
      const float* xp = in + ((size_t)bg * T_ + t) * F_ + l * 8;
      float*       ap = attns + ((size_t)bg * T_ + t) * F_ + l * 8;
      float4 x0 = *(const float4*)xp;
      float4 x1 = *(const float4*)(xp + 4);
      if (t + 1 < T_) __builtin_prefetch(xp + F_, 0, 0);   // next step's x_t -> global_prefetch_b8
      v8bf w8;
      w8[0] = (__bf16)(areg[0] * x0.x); w8[1] = (__bf16)(areg[1] * x0.y);
      w8[2] = (__bf16)(areg[2] * x0.z); w8[3] = (__bf16)(areg[3] * x0.w);
      w8[4] = (__bf16)(areg[4] * x1.x); w8[5] = (__bf16)(areg[5] * x1.y);
      w8[6] = (__bf16)(areg[6] * x1.z); w8[7] = (__bf16)(areg[7] * x1.w);
      *(v8bf*)&wib[b][l * 8] = w8;
      *(float4*)ap       = make_float4(areg[0], areg[1], areg[2], areg[3]);
      *(float4*)(ap + 4) = make_float4(areg[4], areg[5], areg[6], areg[7]);
    }
    __syncthreads();

    // ---- phase 2: gates = wi @ w_ih^T + h @ w_hh^T + bias via WMMA bf16 ----
    // kt-outer / nt-inner with 8 independent accumulator chains AND explicit
    // double-buffered B fragments: loads for k-step kt+1 are issued into the
    // ping-pong buffer before the 8 WMMAs of k-step kt, so a full k-step of
    // matrix work covers the L2 latency (waits become staggered, not 0x0).
    {
      // Wave-local weight bases. Laundered once per timestep: LICM cannot hoist
      // the (t-invariant) fragment loads out of the recurrence loop and spill
      // them to scratch; recast through uintptr_t to address_space(1) so they
      // stay on the GLOBAL path. All 96 fragments are immediate offsets
      // (< 24-bit) from these two bases.
      const __bf16* bw0 = wih + (size_t)(wv * 32) * 512 + ln * 16;
      const __bf16* bh0 = whh + (size_t)(wv * 64) * 512 + ln * 16;
      asm volatile("" : "+v"(bw0), "+v"(bh0));
      gbp bw = (gbp)(unsigned long long)(uintptr_t)bw0;
      gbp bh = (gbp)(unsigned long long)(uintptr_t)bh0;

      // fragment address for (kt, nt); kt 0..3 -> w_ih, kt 4..11 -> w_hh
      auto baddr = [&](int kt, int nt) -> gbp {
        return (kt < 4) ? (bw + (size_t)(nt * 4 + kt) * 512)
                        : (bh + (size_t)(nt * 8 + (kt - 4)) * 512);
      };

      v8f   acc[8] = {};
      v16bf Bb[2][8];

      // prologue: k-step 0 fragments
#pragma unroll
      for (int nt = 0; nt < 8; ++nt) {
        gbp p = baddr(0, nt);
        v8bf lo = *(gv8p)p;
        v8bf hi = *(gv8p)(p + 8);
#pragma unroll
        for (int i = 0; i < 8; ++i) { Bb[0][nt][i] = lo[i]; Bb[0][nt][i + 8] = hi[i]; }
      }

#pragma unroll
      for (int kt = 0; kt < 12; ++kt) {
        const int cur = kt & 1, nxt = cur ^ 1;
        // prefetch next k-step's 8 B fragments (independent of this k-step's WMMAs)
        if (kt < 11) {
#pragma unroll
          for (int nt = 0; nt < 8; ++nt) {
            gbp p = baddr(kt + 1, nt);
            v8bf lo = *(gv8p)p;
            v8bf hi = *(gv8p)(p + 8);
#pragma unroll
            for (int i = 0; i < 8; ++i) { Bb[nxt][nt][i] = lo[i]; Bb[nxt][nt][i + 8] = hi[i]; }
          }
        }
        // A fragment (contiguous k order, not AMD's table order: lanes 0-15 hold K {0..15},
        // lanes 16-31 hold K {16..31}, matching packed B); two ds_load_b128 per fragment.
        const __bf16* ap_ = (kt < 4) ? &wib[row][kt * 32 + sel * 16]
                                     : &hb[row][(kt - 4) * 32 + sel * 16];
        v8bf alo = *(const v8bf*)ap_;
        v8bf ahi = *(const v8bf*)(ap_ + 8);
        v16bf Af;
#pragma unroll
        for (int i = 0; i < 8; ++i) { Af[i] = alo[i]; Af[i + 8] = ahi[i]; }

#pragma unroll
        for (int nt = 0; nt < 8; ++nt) {
          acc[nt] = __builtin_amdgcn_wmma_f32_16x16x32_bf16(false, Af, false, Bb[cur][nt],
                                                            (short)0, acc[nt], false, false);
        }
      }

      // C/D layout: VGPR r, lanes 0-15 -> M=r, lanes 16-31 -> M=8+r; N = lane%16
#pragma unroll
      for (int nt = 0; nt < 8; ++nt) {
        const int n = (wv * 8 + nt) * 16 + row;
        const float bb = biasr[nt];
#pragma unroll
        for (int r = 0; r < 8; ++r) gates[r + 8 * sel][n] = acc[nt][r] + bb;
      }
    }
    __syncthreads();

    // ---- phase 3: LSTM pointwise update; h -> LDS bf16 + enc out ----
    {
      float* ep = enc + ((size_t)bg * T_ + t) * H_;
#pragma unroll
      for (int i = 0; i < 16; ++i) {
        int j = l + 16 * i;
        float gi = gates[b][j];
        float gf = gates[b][j + 256];
        float gg = gates[b][j + 512];
        float go = gates[b][j + 768];
        float cn = sigf(gf) * creg[i] + sigf(gi) * tanhf(gg);
        float hn = sigf(go) * tanhf(cn);
        creg[i] = cn;
        hb[b][j] = (__bf16)hn;
        ep[j] = hn;
      }
    }
    __syncthreads();
  }
}

extern "C" void kernel_launch(void* const* d_in, const int* in_sizes, int n_in,
                              void* d_out, int out_size, void* d_ws, size_t ws_size,
                              hipStream_t stream) {
  const float* input  = (const float*)d_in[0];
  const float* attn_w = (const float*)d_in[1];
  // d_in[2] (attn_b) intentionally unused: softmax shift-invariance cancels it.
  const float* w_ih   = (const float*)d_in[3];
  const float* w_hh   = (const float*)d_in[4];
  const float* b_ih   = (const float*)d_in[5];
  const float* b_hh   = (const float*)d_in[6];
  const float* h0     = (const float*)d_in[7];
  const float* c0     = (const float*)d_in[8];

  char*   ws     = (char*)d_ws;
  float*  xsbuf  = (float*)ws;                      // B*F f32        = 256 KB
  __bf16* wih_pk = (__bf16*)(ws + 256 * 1024);      // G*F bf16       = 256 KB
  __bf16* whh_pk = (__bf16*)(ws + 512 * 1024);      // G*H bf16       = 512 KB

  const float* wx = attn_w + 2 * H_;                // attn_w[2H:] (length T)

  k_xscore<<<(B_ * F_) / 256, 256, 0, stream>>>(input, wx, xsbuf);
  k_pack<<<(G_ * F_) / 256, 256, 0, stream>>>(w_ih, wih_pk, F_);
  k_pack<<<(G_ * H_) / 256, 256, 0, stream>>>(w_hh, whh_pk, H_);

  float* attns = (float*)d_out;                     // (B,T,F)
  float* enc   = attns + (size_t)B_ * T_ * F_;      // (B,T,H)

  k_lstm<<<B_ / BM, 256, 0, stream>>>(input, xsbuf, wih_pk, whh_pk,
                                      b_ih, b_hh, h0, c0, attns, enc);
}
